// FullAttention_11373073400136
// MI455X (gfx1250) — compile-verified, matrix operand order checked by reading, not run on hardware
//
#include <hip/hip_runtime.h>
#include <hip/hip_bf16.h>

// ---------------------------------------------------------------------------
// MI455X (gfx1250) full attention forward.
// GEMMs: v_wmma_f32_16x16x32_f16 with double-buffered LDS weight staging via
// async global->LDS copies (ASYNCcnt) and register-pipelined fragments.
// ---------------------------------------------------------------------------

typedef __attribute__((ext_vector_type(16))) _Float16 v16h;
typedef __attribute__((ext_vector_type(8)))  _Float16 v8h;
typedef __attribute__((ext_vector_type(8)))  float    v8f;
typedef __attribute__((ext_vector_type(4)))  int      v4i;

#define N_EMBD 1024
#define N_HEAD 16
#define HDIM   64
#define BATCH  4
#define SEQ    2048
#define ROWS   (BATCH * SEQ)                      // 8192
#define Y_ELEMS   ((size_t)ROWS * N_EMBD)         // 8,388,608
#define AM_ELEMS  ((size_t)BATCH * SEQ * SEQ)     // 16,777,216

#if __has_builtin(__builtin_amdgcn_global_load_async_to_lds_b128)
#define HAVE_ASYNC_LDS 1
#else
#define HAVE_ASYNC_LDS 0
#endif

#define GLOBAL_AS __attribute__((address_space(1)))
#define LOCAL_AS  __attribute__((address_space(3)))

// ---- WMMA helpers ----------------------------------------------------------

__device__ __forceinline__ v8f wmma_f16(v16h a, v16h b, v8f c) {
  return __builtin_amdgcn_wmma_f32_16x16x32_f16(
      /*neg_a=*/false, a, /*neg_b=*/false, b,
      /*c_mod=*/(short)0, c, /*reuse_a=*/false, /*reuse_b=*/false);
}

// Load a 16x32 fragment from a row-major f16 matrix (row stride ld).
// Lane mapping per ISA 16-bit A layout: lanes 0-15 row=lane half=0,
// lanes 16-31 row=lane-16 half=1; K chunks at 8*half and 16+8*half.
__device__ __forceinline__ v16h load_frag(const _Float16* __restrict__ base, int ld) {
  const int lane = threadIdx.x & 31;
  const int row  = lane & 15;
  const int half = lane >> 4;
  const _Float16* p = base + (size_t)row * ld + 8 * half;
  v8h lo = *(const v8h*)(p);
  v8h hi = *(const v8h*)(p + 16);
  v16h r;
#pragma unroll
  for (int i = 0; i < 8; ++i) { r[i] = lo[i]; r[i + 8] = hi[i]; }
  return r;
}

// Async (or fallback) copy of one 16-byte chunk global->LDS.
__device__ __forceinline__ void copy16_g2l(const _Float16* gp, _Float16* lp) {
#if HAVE_ASYNC_LDS
  __builtin_amdgcn_global_load_async_to_lds_b128(
      (GLOBAL_AS v4i*)(gp), (LOCAL_AS v4i*)(lp), 0, 0);
#else
  *(v8h*)lp = *(const v8h*)gp;
#endif
}

__device__ __forceinline__ void async_wait_barrier() {
#if HAVE_ASYNC_LDS
  asm volatile("s_wait_asynccnt 0x0" ::: "memory");
#endif
  __syncthreads();
}

// ---- Utility kernels -------------------------------------------------------

__global__ void cvt_f32_to_f16(const float* __restrict__ src,
                               _Float16* __restrict__ dst, int n) {
  int i = blockIdx.x * blockDim.x + threadIdx.x;
  int stride = gridDim.x * blockDim.x;
  for (; i < n; i += stride) dst[i] = (_Float16)src[i];
}

__global__ void zero_f32(float* __restrict__ p, int n) {
  int i = blockIdx.x * blockDim.x + threadIdx.x;
  int stride = gridDim.x * blockDim.x;
  for (; i < n; i += stride) p[i] = 0.0f;
}

// ---- QKV projection --------------------------------------------------------
// grid: (ROWS/16/4, N_EMBD/64, 3), block: 128 (4 waves).
// B-tiles (64 rows of W x 32 k) are shared by all 4 waves -> staged in LDS by
// async copies, double buffered. A-fragments are register-pipelined.

__global__ void __launch_bounds__(128) gemm_qkv(
    const _Float16* __restrict__ xh,
    const _Float16* __restrict__ wq, const _Float16* __restrict__ wk,
    const _Float16* __restrict__ wv,
    const float* __restrict__ bq, const float* __restrict__ bk,
    const float* __restrict__ bv,
    _Float16* __restrict__ qh, _Float16* __restrict__ kh,
    _Float16* __restrict__ vt) {
  __shared__ _Float16 sB[2][64 * 32];   // 4KB per buffer

  const int tid  = threadIdx.x;
  const int wave = tid >> 5;
  const int lane = tid & 31;
  const int t0 = (blockIdx.x * 4 + wave) * 16;   // token-row tile
  const int o0 = blockIdx.y * 64;                // output-feature strip
  const int which = blockIdx.z;                  // 0=Q 1=K 2=V
  const _Float16* W = (which == 0) ? wq : (which == 1) ? wk : wv;
  const float* bias = (which == 0) ? bq : (which == 1) ? bk : bv;
  const _Float16* Wblk = W + (size_t)o0 * N_EMBD;
  const _Float16* Ablk = xh + (size_t)t0 * N_EMBD;

  // per-thread chunk coordinates within the 64x32 tile (256 x 16B chunks)
  const int row0 = tid >> 2;                 // chunk tid
  const int k80  = (tid & 3) * 8;
  const int row1 = (tid + 128) >> 2;         // chunk tid+128
  const int k81  = ((tid + 128) & 3) * 8;

  // stage one 64x32 weight tile: two straight-line async copies per thread
  auto stage = [&](int buf, int kk) {
    copy16_g2l(Wblk + (size_t)row0 * N_EMBD + kk + k80,
               &sB[buf][row0 * 32 + k80]);
    copy16_g2l(Wblk + (size_t)row1 * N_EMBD + kk + k81,
               &sB[buf][row1 * 32 + k81]);
  };

  v8f acc[4] = {};
  stage(0, 0);
  v16h a_cur = load_frag(Ablk, N_EMBD);
  async_wait_barrier();

#pragma unroll 2
  for (int i = 0; i < N_EMBD / 32; ++i) {
    const int kk = i * 32;
    if (i + 1 < N_EMBD / 32) stage((i + 1) & 1, kk + 32);   // async, in flight
    v16h a_next = a_cur;
    if (i + 1 < N_EMBD / 32) a_next = load_frag(Ablk + kk + 32, N_EMBD);
    // batch all four B fragments from LDS, then fire 4 back-to-back WMMAs
    v16h bf[4];
#pragma unroll
    for (int s = 0; s < 4; ++s) bf[s] = load_frag(&sB[i & 1][16 * s * 32], 32);
#pragma unroll
    for (int s = 0; s < 4; ++s) acc[s] = wmma_f16(a_cur, bf[s], acc[s]);
    a_cur = a_next;
    async_wait_barrier();
  }

  const int n = lane & 15, half = lane >> 4;
  const int bidx = t0 >> 11;     // /2048
  const int tt0  = t0 & 2047;
  const int h    = o0 >> 6;

  if (which < 2) {
    _Float16* dst = (which == 0) ? qh : kh;
#pragma unroll
    for (int s = 0; s < 4; ++s) {
      const int d = 16 * s + n;
      const float bb = bias[o0 + 16 * s + n];
#pragma unroll
      for (int r = 0; r < 8; ++r) {
        const int tt = tt0 + r + 8 * half;
        dst[(((size_t)(bidx * N_HEAD + h) * SEQ) + tt) * HDIM + d] =
            (_Float16)(acc[s][r] + bb);
      }
    }
  } else {  // V: transposed store, 8 contiguous f16 per lane
#pragma unroll
    for (int s = 0; s < 4; ++s) {
      const int d = 16 * s + n;
      const float bb = bias[o0 + 16 * s + n];
      v8h pk;
#pragma unroll
      for (int r = 0; r < 8; ++r) pk[r] = (_Float16)(acc[s][r] + bb);
      *(v8h*)(vt + ((size_t)(bidx * N_HEAD + h) * HDIM + d) * SEQ + tt0 +
              8 * half) = pk;
    }
  }
}

// ---- Attention: scores + softmax + att_mean + P@V --------------------------
// grid: (SEQ/16, N_HEAD, BATCH), block: 256 (8 waves).

__global__ void __launch_bounds__(256) attn_kernel(
    const _Float16* __restrict__ qh, const _Float16* __restrict__ kh,
    const _Float16* __restrict__ vt, const int* __restrict__ mask,
    _Float16* __restrict__ yheads, float* __restrict__ attmean) {
  __shared__ float sS[16][2052];   // 16 query rows x 2048 keys (padded)
  __shared__ float yred[16][64];   // cross-wave reduction for y tile
  __shared__ float sred[16][16];   // softmax partials
  __shared__ float srow[16];       // per-row broadcast

  const int tid  = threadIdx.x;
  const int wave = tid >> 5, lane = tid & 31;
  const int n = lane & 15, half = lane >> 4;
  const int qt = blockIdx.x, h = blockIdx.y, bb = blockIdx.z;
  const int bh = bb * N_HEAD + h;
  const int t0 = qt * 16;

  const _Float16* qbase = qh + ((size_t)bh * SEQ + t0) * HDIM;
  const v16h aq0 = load_frag(qbase, HDIM);
  const v16h aq1 = load_frag(qbase + 32, HDIM);

  // ---- scores: S = (Q K^T) * 1/sqrt(64), mask applied; K frags pipelined ----
  const _Float16* kbase = kh + (size_t)bh * SEQ * HDIM;
  {
    v16h b0 = load_frag(kbase + (size_t)(wave * 16) * HDIM, HDIM);
    v16h b1 = load_frag(kbase + (size_t)(wave * 16) * HDIM + 32, HDIM);
    for (int j = wave; j < SEQ / 16; j += 8) {
      v16h nb0 = b0, nb1 = b1;
      if (j + 8 < SEQ / 16) {
        nb0 = load_frag(kbase + (size_t)((j + 8) * 16) * HDIM, HDIM);
        nb1 = load_frag(kbase + (size_t)((j + 8) * 16) * HDIM + 32, HDIM);
      }
      v8f acc = {};
      acc = wmma_f16(aq0, b0, acc);
      acc = wmma_f16(aq1, b1, acc);
      const int kcol = j * 16 + n;
      const int mk = mask[bb * SEQ + kcol];
#pragma unroll
      for (int r = 0; r < 8; ++r) {
        const float s = (mk == 0) ? -3.4028235e38f : acc[r] * 0.125f;
        sS[r + 8 * half][kcol] = s;
      }
      b0 = nb0; b1 = nb1;
    }
  }
  __syncthreads();

  // ---- softmax (16 threads per query row) -----------------------------------
  {
    const int row = tid >> 4, sub = tid & 15;
    float mx = -3.4028235e38f;
    for (int i = sub; i < SEQ; i += 16) mx = fmaxf(mx, sS[row][i]);
    sred[row][sub] = mx;
    __syncthreads();
    if (sub == 0) {
      float m2 = sred[row][0];
#pragma unroll
      for (int i = 1; i < 16; ++i) m2 = fmaxf(m2, sred[row][i]);
      srow[row] = m2;
    }
    __syncthreads();
    mx = srow[row];
    float sum = 0.0f;
    for (int i = sub; i < SEQ; i += 16) {
      const float e = __expf(sS[row][i] - mx);
      sS[row][i] = e;
      sum += e;
    }
    __syncthreads();
    sred[row][sub] = sum;
    __syncthreads();
    if (sub == 0) {
      float s2 = 0.0f;
#pragma unroll
      for (int i = 0; i < 16; ++i) s2 += sred[row][i];
      srow[row] = 1.0f / s2;
    }
    __syncthreads();
    const float inv = srow[row];
    float* amrow = attmean + ((size_t)bb * SEQ + t0 + row) * SEQ;
    for (int i = sub; i < SEQ; i += 16) {
      const float p = sS[row][i] * inv;
      sS[row][i] = p;
      atomicAdd(amrow + i, p * (1.0f / N_HEAD));  // mean over heads
    }
  }

  for (int i = tid; i < 16 * 64; i += 256) ((float*)yred)[i] = 0.0f;
  __syncthreads();

  // ---- y = P @ V (each wave owns a 256-key slice); V frags pipelined --------
  const _Float16* vbase = vt + (size_t)bh * HDIM * SEQ;
  {
    v8f acc[4] = {};
    const int m = n;
    const int kk0 = wave * 256;
    v16h bf_cur[4];
#pragma unroll
    for (int s = 0; s < 4; ++s)
      bf_cur[s] = load_frag(vbase + (size_t)(16 * s) * SEQ + kk0, SEQ);
    for (int kk = kk0; kk < kk0 + 256; kk += 32) {
      v16h bf_next[4];
#pragma unroll
      for (int s = 0; s < 4; ++s) bf_next[s] = bf_cur[s];
      if (kk + 32 < kk0 + 256) {
#pragma unroll
        for (int s = 0; s < 4; ++s)
          bf_next[s] = load_frag(vbase + (size_t)(16 * s) * SEQ + kk + 32, SEQ);
      }
      v16h ap;
#pragma unroll
      for (int i = 0; i < 8; ++i) {
        ap[i]     = (_Float16)sS[m][kk + 8 * half + i];
        ap[i + 8] = (_Float16)sS[m][kk + 16 + 8 * half + i];
      }
#pragma unroll
      for (int s = 0; s < 4; ++s) acc[s] = wmma_f16(ap, bf_cur[s], acc[s]);
#pragma unroll
      for (int s = 0; s < 4; ++s) bf_cur[s] = bf_next[s];
    }
#pragma unroll
    for (int s = 0; s < 4; ++s)
#pragma unroll
      for (int r = 0; r < 8; ++r)
        atomicAdd(&yred[r + 8 * half][16 * s + n], acc[s][r]);  // ds_add_f32
  }
  __syncthreads();

  for (int i = tid; i < 16 * 64; i += 256) {
    const int r = i >> 6, d = i & 63;
    yheads[((size_t)bb * SEQ + t0 + r) * N_EMBD + h * HDIM + d] =
        (_Float16)yred[r][d];
  }
}

// ---- Output projection: y = yheads @ Wp^T + bp (f32 out) -------------------
// grid: (ROWS/16/4, N_EMBD/64), block: 128. Same LDS staging as gemm_qkv.

__global__ void __launch_bounds__(128) gemm_proj(
    const _Float16* __restrict__ yh, const _Float16* __restrict__ wp,
    const float* __restrict__ bp, float* __restrict__ out) {
  __shared__ _Float16 sB[2][64 * 32];

  const int tid  = threadIdx.x;
  const int wave = tid >> 5;
  const int lane = tid & 31;
  const int t0 = (blockIdx.x * 4 + wave) * 16;
  const int o0 = blockIdx.y * 64;
  const _Float16* Wblk = wp + (size_t)o0 * N_EMBD;
  const _Float16* Ablk = yh + (size_t)t0 * N_EMBD;

  const int row0 = tid >> 2;
  const int k80  = (tid & 3) * 8;
  const int row1 = (tid + 128) >> 2;
  const int k81  = ((tid + 128) & 3) * 8;

  auto stage = [&](int buf, int kk) {
    copy16_g2l(Wblk + (size_t)row0 * N_EMBD + kk + k80,
               &sB[buf][row0 * 32 + k80]);
    copy16_g2l(Wblk + (size_t)row1 * N_EMBD + kk + k81,
               &sB[buf][row1 * 32 + k81]);
  };

  v8f acc[4] = {};
  stage(0, 0);
  v16h a_cur = load_frag(Ablk, N_EMBD);
  async_wait_barrier();

#pragma unroll 2
  for (int i = 0; i < N_EMBD / 32; ++i) {
    const int kk = i * 32;
    if (i + 1 < N_EMBD / 32) stage((i + 1) & 1, kk + 32);
    v16h a_next = a_cur;
    if (i + 1 < N_EMBD / 32) a_next = load_frag(Ablk + kk + 32, N_EMBD);
    v16h bf[4];
#pragma unroll
    for (int s = 0; s < 4; ++s) bf[s] = load_frag(&sB[i & 1][16 * s * 32], 32);
#pragma unroll
    for (int s = 0; s < 4; ++s) acc[s] = wmma_f16(a_cur, bf[s], acc[s]);
    a_cur = a_next;
    async_wait_barrier();
  }

  const int n = lane & 15, half = lane >> 4;
#pragma unroll
  for (int s = 0; s < 4; ++s) {
    const float bb = bp[o0 + 16 * s + n];
#pragma unroll
    for (int r = 0; r < 8; ++r)
      out[(size_t)(t0 + r + 8 * half) * N_EMBD + o0 + 16 * s + n] =
          acc[s][r] + bb;
  }
}

// ---------------------------------------------------------------------------

extern "C" void kernel_launch(void* const* d_in, const int* in_sizes, int n_in,
                              void* d_out, int out_size, void* d_ws,
                              size_t ws_size, hipStream_t stream) {
  (void)in_sizes; (void)n_in; (void)out_size; (void)ws_size;

  const float* x    = (const float*)d_in[0];
  const int*   mask = (const int*)d_in[1];
  const float* Wq   = (const float*)d_in[2];
  const float* bq   = (const float*)d_in[3];
  const float* Wk   = (const float*)d_in[4];
  const float* bk   = (const float*)d_in[5];
  const float* Wv   = (const float*)d_in[6];
  const float* bv   = (const float*)d_in[7];
  const float* Wp   = (const float*)d_in[8];
  const float* bp   = (const float*)d_in[9];

  float* yout  = (float*)d_out;            // (4,2048,1024)
  float* amean = yout + Y_ELEMS;           // (4,2048,2048)

  _Float16* ws  = (_Float16*)d_ws;
  _Float16* xh  = ws;
  _Float16* wqh = xh  + Y_ELEMS;
  _Float16* wkh = wqh + (size_t)N_EMBD * N_EMBD;
  _Float16* wvh = wkh + (size_t)N_EMBD * N_EMBD;
  _Float16* wph = wvh + (size_t)N_EMBD * N_EMBD;
  _Float16* qh  = wph + (size_t)N_EMBD * N_EMBD;
  _Float16* kh  = qh  + Y_ELEMS;
  _Float16* vt  = kh  + Y_ELEMS;
  _Float16* yh  = vt  + Y_ELEMS;

  cvt_f32_to_f16<<<1024, 256, 0, stream>>>(x,  xh,  (int)Y_ELEMS);
  cvt_f32_to_f16<<<512, 256, 0, stream>>>(Wq, wqh, N_EMBD * N_EMBD);
  cvt_f32_to_f16<<<512, 256, 0, stream>>>(Wk, wkh, N_EMBD * N_EMBD);
  cvt_f32_to_f16<<<512, 256, 0, stream>>>(Wv, wvh, N_EMBD * N_EMBD);
  cvt_f32_to_f16<<<512, 256, 0, stream>>>(Wp, wph, N_EMBD * N_EMBD);

  zero_f32<<<2048, 256, 0, stream>>>(amean, (int)AM_ELEMS);

  gemm_qkv<<<dim3(ROWS / 16 / 4, N_EMBD / 64, 3), 128, 0, stream>>>(
      xh, wqh, wkh, wvh, bq, bk, bv, qh, kh, vt);

  attn_kernel<<<dim3(SEQ / 16, N_HEAD, BATCH), 256, 0, stream>>>(
      qh, kh, vt, mask, yh, amean);

  gemm_proj<<<dim3(ROWS / 16 / 4, N_EMBD / 64), 128, 0, stream>>>(
      yh, wph, bp, yout);
}
